// GatedQGNN_66022237274797
// MI455X (gfx1250) — compile-verified, matrix operand order checked by reading, not run on hardware
//
#include <hip/hip_runtime.h>
#include <stdint.h>

#define BB 8
#define NN 2048
#define FF 128
#define HH 128
#define NSTEP 4
#define NCLS 32

typedef __attribute__((ext_vector_type(16))) __bf16        v16bf;
typedef __attribute__((ext_vector_type(8)))  float         v8f;
typedef __attribute__((ext_vector_type(4)))  unsigned int  u32x4;
typedef __attribute__((ext_vector_type(4)))  float         f32x4;

union Frag16 { u32x4 q[2]; v16bf v; };
union Pack8  { unsigned short s[8]; u32x4 q; };

__device__ __forceinline__ unsigned short f2bf(float f) {
  union { float f; unsigned u; } v; v.f = f;
  unsigned r = v.u + 0x7FFFu + ((v.u >> 16) & 1u);   // round-to-nearest-even
  return (unsigned short)(r >> 16);
}

__device__ __forceinline__ float sigm(float v) { return 1.0f / (1.0f + __expf(-v)); }

__device__ __forceinline__ v8f zero8() {
  v8f z;
#pragma unroll
  for (int i = 0; i < 8; ++i) z[i] = 0.0f;
  return z;
}

// bf16 WMMA fragment loader for row-major data (A: rows=M, B: rows=N of a
// transposed weight).  CDNA5 16-bit A layout: lanes 0-15 hold row M=lane,
// K = {0..7,16..23}; lanes 16-31 hold K = {8..15,24..31}.  Two 16-byte loads.
__device__ __forceinline__ v16bf load_frag(const unsigned short* base, int ld,
                                           int rc0, int kb, int lane) {
  int r  = rc0 + (lane & 15);
  int ko = kb + ((lane & 16) ? 8 : 0);
  const unsigned short* p = base + r * ld + ko;
  Frag16 f;
  f.q[0] = *(const u32x4*)(p);
  f.q[1] = *(const u32x4*)(p + 16);
  return f.v;
}

#define WMMA_BF16(A, Bm, Cacc) \
  __builtin_amdgcn_wmma_f32_16x16x32_bf16(false, (A), false, (Bm), (short)0, (Cacc), false, false)

// Issue one 64x128 bf16 adjacency tile (16 KB) as per-lane async copies into
// LDS: 1024 16-byte chunks, 4 per thread, tracked on ASYNCcnt.
__device__ __forceinline__ void stage_adj_async(const unsigned short* adjb,
                                                unsigned short* dst_lds,
                                                int k0, int tid) {
  for (int c = tid; c < 1024; c += 256) {
    int r  = c >> 4;
    int kc = (c & 15) * 8;
    unsigned lds_off = (unsigned)(size_t)&dst_lds[r * 128 + kc];
    const unsigned short* g = adjb + (size_t)r * NN + k0 + kc;
    asm volatile("global_load_async_to_lds_b128 %0, %1, off"
                 :: "v"(lds_off), "v"(g) : "memory");
  }
}

// ---------------------------------------------------------------------------
// Build the 8 Hamilton matrices, TRANSPOSED ([out n][in k]) in bf16.
// H[k=bi*32+rr][n=bj*32+cc] = sign[bj][bi] * Kq[rr][comp[bj][bi]*32 + cc]
// ---------------------------------------------------------------------------
__global__ void k_build_hT(const float* e, const float* z0, const float* z1,
                           const float* r0, const float* r1, const float* h0,
                           const float* h1, const float* ln,
                           unsigned short* __restrict__ hT) {
  const float* srcs[8] = {e, z0, z1, r0, r1, h0, h1, ln};
  const int   ctab[4][4] = {{0,1,2,3},{1,0,3,2},{2,3,0,1},{3,2,1,0}};
  const float stab[4][4] = {{1.f,-1.f,-1.f,-1.f},{1.f,1.f,-1.f,1.f},
                            {1.f,1.f,1.f,-1.f},{1.f,-1.f,1.f,1.f}};
  const float* Kq = srcs[blockIdx.x];
  unsigned short* out = hT + (size_t)blockIdx.x * HH * HH;
  for (int idx = threadIdx.x; idx < HH * HH; idx += blockDim.x) {
    int n = idx >> 7, k = idx & 127;
    int bj = n >> 5, cc = n & 31, bi = k >> 5, rr = k & 31;
    float v = stab[bj][bi] * Kq[rr * HH + ctab[bj][bi] * 32 + cc];
    out[idx] = f2bf(v);   // out[n*128 + k] = H[k][n]
  }
}

// adj f32 -> bf16 (one-time per launch); 4 elems/thread
__global__ void k_cvt_bf16(const float* __restrict__ src,
                           unsigned short* __restrict__ dst) {
  size_t i = ((size_t)blockIdx.x * blockDim.x + threadIdx.x) * 4;
  f32x4 v = *(const f32x4*)(src + i);
  union { unsigned short s[4]; unsigned long long u; } o;
  o.s[0] = f2bf(v.x); o.s[1] = f2bf(v.y); o.s[2] = f2bf(v.z); o.s[3] = f2bf(v.w);
  *(unsigned long long*)(dst + i) = o.u;
}

// ---------------------------------------------------------------------------
// Encode: x = (inputs @ Hemb) * mask ; writes x (f32 row-major) and
// xT (bf16, [b][hid][node]) for the SpMM B-operand.
// ---------------------------------------------------------------------------
__global__ void k_encode(const float* __restrict__ inputs,
                         const float* __restrict__ mask,
                         const unsigned short* __restrict__ hT,
                         float* __restrict__ x,
                         unsigned short* __restrict__ xT) {
  __shared__ unsigned short s_in[64 * FF];
  __shared__ float s_m[64];
  int blk = blockIdx.x, b = blk >> 5, m0 = (blk & 31) * 64;
  const float* src = inputs + ((size_t)b * NN + m0) * FF;
  for (int i = threadIdx.x; i < 64 * FF; i += 256) s_in[i] = f2bf(src[i]);
  if (threadIdx.x < 64) s_m[threadIdx.x] = mask[(size_t)b * NN + m0 + threadIdx.x];
  __syncthreads();

  int lane = threadIdx.x & 31, wave = threadIdx.x >> 5;
  int rt = wave & 3, ch = wave >> 2;
  v8f acc[4];
#pragma unroll
  for (int j = 0; j < 4; ++j) acc[j] = zero8();

#pragma unroll
  for (int kb = 0; kb < FF; kb += 32) {
    v16bf A = load_frag(s_in, FF, rt * 16, kb, lane);
#pragma unroll
    for (int j = 0; j < 4; ++j)
      acc[j] = WMMA_BF16(A, load_frag(hT, HH, (ch * 4 + j) * 16, kb, lane), acc[j]);
  }

  float* xb = x + ((size_t)b * NN + m0) * HH;
  unsigned short* xTb = xT + (size_t)b * HH * NN;
  int mb = rt * 16 + ((lane & 16) ? 8 : 0);
#pragma unroll
  for (int j = 0; j < 4; ++j) {
    int n = (ch * 4 + j) * 16 + (lane & 15);
    Pack8 pk;
#pragma unroll
    for (int i = 0; i < 8; ++i) {
      float v = acc[j][i] * s_m[mb + i];
      xb[(mb + i) * HH + n] = v;
      pk.s[i] = f2bf(v);
    }
    *(u32x4*)(xTb + (size_t)n * NN + m0 + mb) = pk.q;
  }
}

// ---------------------------------------------------------------------------
// SpMM:  a = adj @ x   (bf16 inputs, f32 accumulate).  Dominant HBM consumer:
// adj tiles stream via CDNA5 async-to-LDS copies, double-buffered so chunk
// i+1's DMA overlaps chunk i's WMMA (wait "asynccnt <= 4" releases only the
// oldest in-order batch).  x comes from the transposed bf16 copy (L2-hot).
// ---------------------------------------------------------------------------
__global__ void k_spmm(const unsigned short* __restrict__ adj_bf,
                       const unsigned short* __restrict__ xT,
                       float* __restrict__ a) {
  __shared__ unsigned short s_adj[2][64 * 128];   // 2 x 16 KB double buffer
  int blk = blockIdx.x, b = blk >> 5, m0 = (blk & 31) * 64;
  const unsigned short* adjb = adj_bf + ((size_t)b * NN + m0) * NN;
  const unsigned short* xTb  = xT + (size_t)b * HH * NN;
  int lane = threadIdx.x & 31, wave = threadIdx.x >> 5;
  int rt = wave & 3, ch = wave >> 2;
  v8f acc[4];
#pragma unroll
  for (int j = 0; j < 4; ++j) acc[j] = zero8();

  stage_adj_async(adjb, &s_adj[0][0], 0, threadIdx.x);

  for (int k0 = 0; k0 < NN; k0 += 128) {
    int cur = (k0 >> 7) & 1;
    if (k0 + 128 < NN) {
      stage_adj_async(adjb, &s_adj[cur ^ 1][0], k0 + 128, threadIdx.x);
      // 4 newest (next chunk) may remain in flight; 4 oldest (this chunk)
      // must have landed -- async loads complete in order.
      asm volatile("s_wait_asynccnt 4" ::: "memory");
    } else {
      asm volatile("s_wait_asynccnt 0" ::: "memory");
    }
    __syncthreads();   // publish this chunk's LDS writes to all waves

#pragma unroll
    for (int kk = 0; kk < 128; kk += 32) {
      v16bf A = load_frag(&s_adj[cur][0], 128, rt * 16, kk, lane);
#pragma unroll
      for (int j = 0; j < 4; ++j)
        acc[j] = WMMA_BF16(A, load_frag(xTb, NN, (ch * 4 + j) * 16, k0 + kk, lane), acc[j]);
    }
    __syncthreads();   // all waves done with buf `cur` before it is restaged
  }

  float* ab = a + ((size_t)b * NN + m0) * HH;
  int mb = rt * 16 + ((lane & 16) ? 8 : 0);
#pragma unroll
  for (int j = 0; j < 4; ++j) {
    int n = (ch * 4 + j) * 16 + (lane & 15);
#pragma unroll
    for (int i = 0; i < 8; ++i) ab[(mb + i) * HH + n] = acc[j][i];
  }
}

// ---------------------------------------------------------------------------
// Fused GRU gate update: z,r = sigmoid(a@Hz0 + x@Hz1 / a@Hr0 + x@Hr1),
// h = relu(a@Hh0 + (r*x)@Hh1),  x = (h*z + x*(1-z)) * mask.
// Each block owns 64 rows; Hamilton weights streamed from L2 as B-fragments.
// ---------------------------------------------------------------------------
__global__ void k_gate(const float* __restrict__ a, float* __restrict__ x,
                       unsigned short* __restrict__ xT,
                       const unsigned short* __restrict__ hT,
                       const float* __restrict__ mask) {
  __shared__ unsigned short s_a[64 * HH];
  __shared__ unsigned short s_x[64 * HH];
  __shared__ unsigned short s_rx[64 * HH];
  __shared__ float s_xf[64 * HH];
  __shared__ float s_m[64];

  int blk = blockIdx.x, b = blk >> 5, m0 = (blk & 31) * 64;
  const float* ab = a + ((size_t)b * NN + m0) * HH;
  float* xb = x + ((size_t)b * NN + m0) * HH;
  unsigned short* xTb = xT + (size_t)b * HH * NN;

  for (int i = threadIdx.x; i < 64 * HH; i += 256) {
    float av = ab[i], xv = xb[i];
    s_a[i] = f2bf(av);
    s_xf[i] = xv;
    s_x[i] = f2bf(xv);
  }
  if (threadIdx.x < 64) s_m[threadIdx.x] = mask[(size_t)b * NN + m0 + threadIdx.x];
  __syncthreads();

  const unsigned short* Hz0 = hT + 1 * HH * HH;
  const unsigned short* Hz1 = hT + 2 * HH * HH;
  const unsigned short* Hr0 = hT + 3 * HH * HH;
  const unsigned short* Hr1 = hT + 4 * HH * HH;
  const unsigned short* Hh0 = hT + 5 * HH * HH;
  const unsigned short* Hh1 = hT + 6 * HH * HH;

  int lane = threadIdx.x & 31, wave = threadIdx.x >> 5;
  int rt = wave & 3, ch = wave >> 2;
  v8f zacc[4], racc[4];
#pragma unroll
  for (int j = 0; j < 4; ++j) { zacc[j] = zero8(); racc[j] = zero8(); }

#pragma unroll
  for (int kb = 0; kb < HH; kb += 32) {
    v16bf Aa = load_frag(s_a, HH, rt * 16, kb, lane);
    v16bf Ax = load_frag(s_x, HH, rt * 16, kb, lane);
#pragma unroll
    for (int j = 0; j < 4; ++j) {
      int c0 = (ch * 4 + j) * 16;
      zacc[j] = WMMA_BF16(Aa, load_frag(Hz0, HH, c0, kb, lane), zacc[j]);
      zacc[j] = WMMA_BF16(Ax, load_frag(Hz1, HH, c0, kb, lane), zacc[j]);
      racc[j] = WMMA_BF16(Aa, load_frag(Hr0, HH, c0, kb, lane), racc[j]);
      racc[j] = WMMA_BF16(Ax, load_frag(Hr1, HH, c0, kb, lane), racc[j]);
    }
  }

  int mb = rt * 16 + ((lane & 16) ? 8 : 0);
#pragma unroll
  for (int j = 0; j < 4; ++j) {
    int n = (ch * 4 + j) * 16 + (lane & 15);
#pragma unroll
    for (int i = 0; i < 8; ++i) {
      float zv = sigm(zacc[j][i]);
      float rv = sigm(racc[j][i]);
      zacc[j][i] = zv;                                  // keep z for combine
      s_rx[(mb + i) * HH + n] = f2bf(rv * s_xf[(mb + i) * HH + n]);
    }
  }
  __syncthreads();

  v8f hacc[4];
#pragma unroll
  for (int j = 0; j < 4; ++j) hacc[j] = zero8();
#pragma unroll
  for (int kb = 0; kb < HH; kb += 32) {
    v16bf Aa = load_frag(s_a, HH, rt * 16, kb, lane);
    v16bf Ar = load_frag(s_rx, HH, rt * 16, kb, lane);
#pragma unroll
    for (int j = 0; j < 4; ++j) {
      int c0 = (ch * 4 + j) * 16;
      hacc[j] = WMMA_BF16(Aa, load_frag(Hh0, HH, c0, kb, lane), hacc[j]);
      hacc[j] = WMMA_BF16(Ar, load_frag(Hh1, HH, c0, kb, lane), hacc[j]);
    }
  }

#pragma unroll
  for (int j = 0; j < 4; ++j) {
    int n = (ch * 4 + j) * 16 + (lane & 15);
    Pack8 pk;
#pragma unroll
    for (int i = 0; i < 8; ++i) {
      float hv = fmaxf(hacc[j][i], 0.0f);
      float zv = zacc[j][i];
      float xv = s_xf[(mb + i) * HH + n];
      float xn = (hv * zv + xv * (1.0f - zv)) * s_m[mb + i];
      xb[(mb + i) * HH + n] = xn;
      pk.s[i] = f2bf(xn);
    }
    *(u32x4*)(xTb + (size_t)n * NN + m0 + mb) = pk.q;
  }
}

// ---------------------------------------------------------------------------
// soft_att = sigmoid(x @ W_att + b_att);  x2 = soft_att * relu(x @ Hln) * mask
// ---------------------------------------------------------------------------
__global__ void k_attn(const float* __restrict__ x,
                       const unsigned short* __restrict__ hT,
                       const float* __restrict__ W_att,
                       const float* __restrict__ b_att,
                       const float* __restrict__ mask,
                       float* __restrict__ x2) {
  __shared__ unsigned short s_x[64 * HH];
  __shared__ float s_xf[64 * HH];
  __shared__ float s_w[HH];
  __shared__ float s_att[64];
  __shared__ float s_m[64];

  int blk = blockIdx.x, b = blk >> 5, m0 = (blk & 31) * 64;
  const float* xbin = x + ((size_t)b * NN + m0) * HH;
  for (int i = threadIdx.x; i < 64 * HH; i += 256) {
    float v = xbin[i];
    s_xf[i] = v;
    s_x[i] = f2bf(v);
  }
  if (threadIdx.x < HH) s_w[threadIdx.x] = W_att[threadIdx.x];
  if (threadIdx.x < 64) s_m[threadIdx.x] = mask[(size_t)b * NN + m0 + threadIdx.x];
  __syncthreads();

  if (threadIdx.x < 64) {
    float d = 0.0f;
    for (int k = 0; k < HH; ++k) d += s_xf[threadIdx.x * HH + k] * s_w[k];
    s_att[threadIdx.x] = sigm(d + b_att[0]);
  }
  __syncthreads();

  const unsigned short* Hl = hT + 7 * HH * HH;
  int lane = threadIdx.x & 31, wave = threadIdx.x >> 5;
  int rt = wave & 3, ch = wave >> 2;
  v8f acc[4];
#pragma unroll
  for (int j = 0; j < 4; ++j) acc[j] = zero8();
#pragma unroll
  for (int kb = 0; kb < HH; kb += 32) {
    v16bf A = load_frag(s_x, HH, rt * 16, kb, lane);
#pragma unroll
    for (int j = 0; j < 4; ++j)
      acc[j] = WMMA_BF16(A, load_frag(Hl, HH, (ch * 4 + j) * 16, kb, lane), acc[j]);
  }

  float* ob = x2 + ((size_t)b * NN + m0) * HH;
  int mb = rt * 16 + ((lane & 16) ? 8 : 0);
#pragma unroll
  for (int j = 0; j < 4; ++j) {
    int n = (ch * 4 + j) * 16 + (lane & 15);
#pragma unroll
    for (int i = 0; i < 8; ++i) {
      float v = fmaxf(acc[j][i], 0.0f) * s_att[mb + i] * s_m[mb + i];
      ob[(mb + i) * HH + n] = v;
    }
  }
}

// graph_emb[b][h] = sum_n(x2) * max_n(x2)
__global__ void k_stats(const float* __restrict__ x2, float* __restrict__ gemb) {
  int b = blockIdx.x;
  int h = threadIdx.x & 127, half = threadIdx.x >> 7;
  const float* p = x2 + (size_t)b * NN * HH;
  float s = 0.0f, mx = -1e30f;
  for (int n = half * 1024; n < half * 1024 + 1024; ++n) {
    float v = p[(size_t)n * HH + h];
    s += v;
    mx = fmaxf(mx, v);
  }
  __shared__ float ss[256], sm[256];
  ss[threadIdx.x] = s; sm[threadIdx.x] = mx;
  __syncthreads();
  if (half == 0)
    gemb[b * HH + h] = (ss[h] + ss[128 + h]) * fmaxf(sm[h], sm[128 + h]);
}

// out = graph_emb @ W_pred + b_pred   (8x128 @ 128x32 — trivial)
__global__ void k_pred(const float* __restrict__ gemb, const float* __restrict__ Wp,
                       const float* __restrict__ bp, float* __restrict__ out) {
  int t = threadIdx.x, b = t >> 5, c = t & 31;
  float s = bp[c];
  for (int h = 0; h < HH; ++h) s += gemb[b * HH + h] * Wp[h * NCLS + c];
  out[b * NCLS + c] = s;
}

extern "C" void kernel_launch(void* const* d_in, const int* in_sizes, int n_in,
                              void* d_out, int out_size, void* d_ws, size_t ws_size,
                              hipStream_t stream) {
  const float* inputs = (const float*)d_in[0];
  const float* adj    = (const float*)d_in[1];
  const float* mask   = (const float*)d_in[2];
  const float* qemb   = (const float*)d_in[3];
  const float* qz0    = (const float*)d_in[4];
  const float* qz1    = (const float*)d_in[5];
  const float* qr0    = (const float*)d_in[6];
  const float* qr1    = (const float*)d_in[7];
  const float* qh0    = (const float*)d_in[8];
  const float* qh1    = (const float*)d_in[9];
  const float* qln    = (const float*)d_in[10];
  const float* W_att  = (const float*)d_in[11];
  const float* b_att  = (const float*)d_in[12];
  const float* W_pred = (const float*)d_in[13];
  const float* b_pred = (const float*)d_in[14];
  float* out = (float*)d_out;

  char* ws = (char*)d_ws;
  size_t off = 0;
  unsigned short* hT = (unsigned short*)(ws + off); off += (size_t)8 * HH * HH * 2;
  unsigned short* adj_bf = (unsigned short*)(ws + off); off += (size_t)BB * NN * NN * 2;
  float* x = (float*)(ws + off); off += (size_t)BB * NN * HH * 4;
  unsigned short* xT = (unsigned short*)(ws + off); off += (size_t)BB * NN * HH * 2;
  float* abuf = (float*)(ws + off); off += (size_t)BB * NN * HH * 4;   // also reused as x2
  float* gemb = (float*)(ws + off); off += (size_t)BB * HH * 4;
  (void)ws_size; (void)in_sizes; (void)n_in; (void)out_size;

  k_build_hT<<<8, 256, 0, stream>>>(qemb, qz0, qz1, qr0, qr1, qh0, qh1, qln, hT);
  k_cvt_bf16<<<(BB * NN * NN) / (256 * 4), 256, 0, stream>>>(adj, adj_bf);
  k_encode<<<BB * NN / 64, 256, 0, stream>>>(inputs, mask, hT, x, xT);
  for (int s = 0; s < NSTEP; ++s) {
    k_spmm<<<BB * NN / 64, 256, 0, stream>>>(adj_bf, xT, abuf);
    k_gate<<<BB * NN / 64, 256, 0, stream>>>(abuf, x, xT, hT, mask);
  }
  k_attn<<<BB * NN / 64, 256, 0, stream>>>(x, hT, W_att, b_att, mask, abuf);
  k_stats<<<BB, 256, 0, stream>>>(abuf, gemb);
  k_pred<<<1, 256, 0, stream>>>(gemb, W_pred, b_pred, out);
}
